// DampedStyleQuantumLayer_53317724013115
// MI455X (gfx1250) — matrix operand order repaired and checked by reading, not compiled, not run on hardware
//
#include <hip/hip_runtime.h>
#include <hip/hip_bf16.h>
#include <math.h>

typedef __attribute__((ext_vector_type(16))) _Float16 v16h;
typedef __attribute__((ext_vector_type(8)))  _Float16 v8h;
typedef __attribute__((ext_vector_type(8)))  float    v8f;

#define NQ   8
#define NWQ  9          // total qubits
#define NL   4
#define NST  512        // 2^9 amplitudes
#define NCOL 256        // compact columns (qubit 8 fixed at |0>)
#define BT   16         // batch tile per workgroup
#define DIM  256

struct c2 { float x, y; };

// ---------------------------------------------------------------------------
// Kernel A: build the fixed 512x256 unitary U (everything after the data
// encoding layer) by simulating each compact basis column in LDS.
// One block per column, 256 threads, 9-qubit state (512 c2) in LDS.
// ---------------------------------------------------------------------------

__device__ inline void gate1(c2* st, int tid, int k,
                             c2 g00, c2 g01, c2 g10, c2 g11) {
  // each thread owns one (s0, s1) pair; bit k selects the pair
  int s0 = ((tid >> k) << (k + 1)) | (tid & ((1 << k) - 1));
  int s1 = s0 | (1 << k);
  c2 a0 = st[s0], a1 = st[s1];
  c2 r0 = { g00.x*a0.x - g00.y*a0.y + g01.x*a1.x - g01.y*a1.y,
            g00.x*a0.y + g00.y*a0.x + g01.x*a1.y + g01.y*a1.x };
  c2 r1 = { g10.x*a0.x - g10.y*a0.y + g11.x*a1.x - g11.y*a1.y,
            g10.x*a0.y + g10.y*a0.x + g11.x*a1.y + g11.y*a1.x };
  st[s0] = r0; st[s1] = r1;
  __syncthreads();
}

__device__ inline void cnotg(c2* st, int tid, int kc, int kt) {
  if (tid < 128) {
    int lo = kc < kt ? kc : kt;
    int hi = kc < kt ? kt : kc;
    int x0 = tid & ((1 << lo) - 1);
    int x1 = (tid >> lo) & ((1 << (hi - 1 - lo)) - 1);
    int x2 = tid >> (hi - 1);
    int s  = x0 | (x1 << (lo + 1)) | (x2 << (hi + 1)) | (1 << kc); // ctrl=1, tgt=0
    int s1 = s | (1 << kt);
    c2 t0 = st[s]; st[s] = st[s1]; st[s1] = t0;
  }
  __syncthreads();
}

__global__ void __launch_bounds__(256)
build_unitary_kernel(const float* __restrict__ qw,
                     _Float16* __restrict__ Ur,
                     _Float16* __restrict__ Ui) {
  __shared__ c2 st[NST];
  __shared__ float qws[NL * (NQ + 1)];
  int tid = threadIdx.x;
  int col = blockIdx.x;                 // compact column index s' in 0..255
  if (tid < NL * (NQ + 1)) qws[tid] = qw[tid];
  int s_init = col << 1;                // qubit 8 (bit 0) == 0
  st[tid]       = { (tid       == s_init) ? 1.f : 0.f, 0.f };
  st[tid + 256] = { ((tid+256) == s_init) ? 1.f : 0.f, 0.f };
  __syncthreads();

  // layer 0: RX(w0i) then RZ(w0i) on qubit i  (qubit q <-> bit 8-q)
  for (int i = 0; i < NQ; ++i) {
    float hc = 0.5f * qws[i];
    float c = cosf(hc), s = sinf(hc);
    gate1(st, tid, 8 - i, {c,0.f}, {0.f,-s}, {0.f,-s}, {c,0.f});   // RX
    gate1(st, tid, 8 - i, {c,-s}, {0.f,0.f}, {0.f,0.f}, {c,s});    // RZ
  }
  for (int l = 1; l < NL; ++l) {
    for (int i = 0; i < NQ; ++i)
      cnotg(st, tid, 8 - i, 8 - ((i + 1) & 7));                    // CNOT ring
    for (int i = 0; i < NQ; ++i) {
      float hc = 0.5f * qws[l * (NQ + 1) + i];
      float c = cosf(hc), s = sinf(hc);
      gate1(st, tid, 8 - i, {c,0.f}, {-s,0.f}, {s,0.f}, {c,0.f});  // RY
      gate1(st, tid, 8 - i, {c,-s}, {0.f,0.f}, {0.f,0.f}, {c,s});  // RZ
    }
    cnotg(st, tid, 1, 0);                                          // CNOT(q7 -> q8)
    {
      float hc = 0.5f * qws[l * (NQ + 1) + NQ];
      float c = cosf(hc), s = sinf(hc);
      gate1(st, tid, 0, {c,0.f}, {-s,0.f}, {s,0.f}, {c,0.f});      // RY on q8
      gate1(st, tid, 0, {c,-s}, {0.f,0.f}, {0.f,0.f}, {c,s});      // RZ on q8
    }
  }
  // store column (row-major U, K-contiguous for WMMA A fragments)
  Ur[tid * NCOL + col]         = (_Float16)st[tid].x;
  Ui[tid * NCOL + col]         = (_Float16)st[tid].y;
  Ur[(tid + 256) * NCOL + col] = (_Float16)st[tid + 256].x;
  Ui[(tid + 256) * NCOL + col] = (_Float16)st[tid + 256].y;
}

// ---------------------------------------------------------------------------
// Kernel B: fused  tanh-GEMV -> layernorm -> product-state expansion ->
// complex GEMM amp = U * psi  (WMMA f16, f32 accumulate) -> |amp|^2 ->
// signed Z reduction -> out = x + z*Wo^T + bo.
// One workgroup (8 wave32) per 16 batch rows. Each wave owns 64 amp rows.
// ---------------------------------------------------------------------------
__global__ void __launch_bounds__(256)
qlayer_kernel(const float* __restrict__ x,
              const float* __restrict__ Wp,
              const float* __restrict__ bp,
              const float* __restrict__ gmm,
              const float* __restrict__ bta,
              const float* __restrict__ Wo,
              const float* __restrict__ bo,
              const _Float16* __restrict__ Ur,
              const _Float16* __restrict__ Ui,
              float* __restrict__ out) {
  __shared__ float xs[BT][DIM];                                    // 16 KB
  __shared__ float hsm[BT][NQ];
  __shared__ float fac[BT][NQ][4];                                 // f(0),f(1) re/im
  __shared__ _Float16 Pr[BT][NCOL] __attribute__((aligned(32)));   // psi real
  __shared__ _Float16 Pq[BT][NCOL] __attribute__((aligned(32)));   // psi imag
  __shared__ _Float16 Pn[BT][NCOL] __attribute__((aligned(32)));   // -psi imag
  __shared__ float zpart[256][NQ];                                 // 8 KB
  __shared__ float zf[NQ][BT];

  int tid = threadIdx.x;
  int bbase = blockIdx.x * BT;

  // ---- stage x tile ----
  for (int j = tid; j < BT * DIM; j += 256) {
    int b = j >> 8, c = j & 255;
    xs[b][c] = x[(size_t)(bbase + b) * DIM + c];
  }
  __syncthreads();

  // ---- h = tanh(x Wp^T + bp) ----
  if (tid < 128) {
    int b = tid >> 3, q = tid & 7;
    float acc = bp[q];
    for (int c = 0; c < DIM; ++c) acc += xs[b][c] * Wp[q * DIM + c];
    hsm[b][q] = tanhf(acc);
  }
  __syncthreads();

  // ---- layernorm + encoding factors: f(0)=cos*e^{-i t/2}, f(1)=-i sin*e^{+i t/2}
  if (tid < BT) {
    int b = tid;
    float mu = 0.f;
    for (int q = 0; q < NQ; ++q) mu += hsm[b][q];
    mu *= 0.125f;
    float var = 0.f;
    for (int q = 0; q < NQ; ++q) { float d = hsm[b][q] - mu; var += d * d; }
    var *= 0.125f;
    float inv = rsqrtf(var + 1e-5f);
    for (int q = 0; q < NQ; ++q) {
      float hn = (hsm[b][q] - mu) * inv * gmm[q] + bta[q];
      float hc = 0.5f * hn;
      float c = cosf(hc), s = sinf(hc);
      fac[b][q][0] = c * c;  fac[b][q][1] = -c * s;   // bit 0
      fac[b][q][2] = s * s;  fac[b][q][3] = -s * c;   // bit 1
    }
  }
  __syncthreads();

  // ---- expand product state psi (batch-major f16, K contiguous) ----
  {
    int sp = tid;                         // compact state index 0..255
    for (int b = 0; b < BT; ++b) {
      float pr = 1.f, pi = 0.f;
      for (int i = 0; i < NQ; ++i) {
        int bit = (sp >> (7 - i)) & 1;    // qubit i <-> bit 7-i of s'
        float fr = fac[b][i][2 * bit], fi = fac[b][i][2 * bit + 1];
        float nr = pr * fr - pi * fi;
        pi = pr * fi + pi * fr;
        pr = nr;
      }
      Pr[b][sp] = (_Float16)pr;
      Pq[b][sp] = (_Float16)pi;
      Pn[b][sp] = (_Float16)(-pi);
    }
  }
  __syncthreads();

  // ---- complex GEMM: amp[512 x 16] = U[512 x 256] * psi[256 x 16] ----
  int lane = tid & 31;
  int wv   = tid >> 5;
  int nc   = lane & 15;     // output column (batch in tile) / A row in tile
  int lh   = lane >> 4;     // lane half

  v8f vzero = {0.f,0.f,0.f,0.f,0.f,0.f,0.f,0.f};
  v8f accR[4], accI[4];
#pragma unroll
  for (int mt = 0; mt < 4; ++mt) { accR[mt] = vzero; accI[mt] = vzero; }

  for (int ks = 0; ks < 8; ++ks) {
    int k0 = ks * 32;
    // B fragment: lane-half lh holds 16 contiguous K at k0 + lh*16, column nc
    v16h br = *(const v16h*)&Pr[nc][k0 + lh * 16];
    v16h bi = *(const v16h*)&Pq[nc][k0 + lh * 16];
    v16h bn = *(const v16h*)&Pn[nc][k0 + lh * 16];
#pragma unroll
    for (int mt = 0; mt < 4; ++mt) {
      int row = wv * 64 + mt * 16 + nc;   // A row (M = lane&15)
      const _Float16* par = Ur + (size_t)row * NCOL + k0 + lh * 16;
      const _Float16* pai = Ui + (size_t)row * NCOL + k0 + lh * 16;
      union { v16h v; v8h h[2]; } ar, ai;
      ar.h[0] = *(const v8h*)par;         // K = k0+lh*16 .. +7
      ar.h[1] = *(const v8h*)(par + 8);   // K = k0+lh*16+8 .. +15
      ai.h[0] = *(const v8h*)pai;
      ai.h[1] = *(const v8h*)(pai + 8);
      // Cr += Ur*Pr + Ui*(-Pi) ; Ci += Ur*Pi + Ui*Pr
      accR[mt] = __builtin_amdgcn_wmma_f32_16x16x32_f16(false, ar.v, false, br,
                                                        (short)0, accR[mt], false, false);
      accR[mt] = __builtin_amdgcn_wmma_f32_16x16x32_f16(false, ai.v, false, bn,
                                                        (short)0, accR[mt], false, false);
      accI[mt] = __builtin_amdgcn_wmma_f32_16x16x32_f16(false, ar.v, false, bi,
                                                        (short)0, accI[mt], false, false);
      accI[mt] = __builtin_amdgcn_wmma_f32_16x16x32_f16(false, ai.v, false, br,
                                                        (short)0, accI[mt], false, false);
    }
  }

  // ---- probs + signed reduction: z_w = sum_s (+/-) |amp_s|^2 ----
  float z8[NQ];
#pragma unroll
  for (int w = 0; w < NQ; ++w) z8[w] = 0.f;
#pragma unroll
  for (int mt = 0; mt < 4; ++mt) {
#pragma unroll
    for (int v = 0; v < 8; ++v) {
      int row = wv * 64 + mt * 16 + lh * 8 + v;   // D: M = v + 8*lh
      float cr = accR[mt][v], ci = accI[mt][v];
      float p = cr * cr + ci * ci;
#pragma unroll
      for (int w = 0; w < NQ; ++w)
        z8[w] += ((row >> (8 - w)) & 1) ? -p : p; // qubit w <-> bit 8-w
    }
  }
#pragma unroll
  for (int w = 0; w < NQ; ++w) zpart[tid][w] = z8[w];
  __syncthreads();

  if (tid < 128) {
    int c = tid & 15, w = tid >> 4;
    float acc = 0.f;
#pragma unroll
    for (int wi = 0; wi < 8; ++wi)
      acc += zpart[wi * 32 + c][w] + zpart[wi * 32 + 16 + c][w];
    zf[w][c] = acc;
  }
  __syncthreads();

  // ---- epilogue: out = x + z * Wo^T + bo ----
  {
    int d = tid;
    float wo8[NQ];
#pragma unroll
    for (int w = 0; w < NQ; ++w) wo8[w] = Wo[d * NQ + w];
    float bod = bo[d];
    for (int b = 0; b < BT; ++b) {
      float acc = xs[b][d] + bod;
#pragma unroll
      for (int w = 0; w < NQ; ++w) acc += zf[w][b] * wo8[w];
      out[(size_t)(bbase + b) * DIM + d] = acc;
    }
  }
}

// ---------------------------------------------------------------------------

extern "C" void kernel_launch(void* const* d_in, const int* in_sizes, int n_in,
                              void* d_out, int out_size, void* d_ws, size_t ws_size,
                              hipStream_t stream) {
  const float* x   = (const float*)d_in[0];
  const float* Wp  = (const float*)d_in[1];
  const float* bp  = (const float*)d_in[2];
  const float* gmm = (const float*)d_in[3];
  const float* bta = (const float*)d_in[4];
  const float* qw  = (const float*)d_in[5];
  const float* Wo  = (const float*)d_in[6];
  const float* bo  = (const float*)d_in[7];
  float* out = (float*)d_out;

  const int B = in_sizes[0] / DIM;          // 8192

  // workspace: U as f16 real/imag, row-major [512][256]
  size_t ucount = (size_t)NST * NCOL;
  _Float16* Ur = (_Float16*)d_ws;
  _Float16* Ui = Ur + ucount;

  build_unitary_kernel<<<NCOL, 256, 0, stream>>>(qw, Ur, Ui);
  qlayer_kernel<<<B / BT, 256, 0, stream>>>(x, Wp, bp, gmm, bta, Wo, bo, Ur, Ui, out);
}
